// TransformerBlockQuantum_65481071396988
// MI455X (gfx1250) — compile-verified, matrix operand order checked by reading, not run on hardware
//
#include <hip/hip_runtime.h>

// Fused quantum-transformer block for gfx1250 (MI455X).
// Analytic collapse of the 8-qubit circuit to 8 cosines, then
// 3 WMMA f16 GEMMs (f32 accum) + 2 LayerNorms, fully fused, LDS-staged.

typedef _Float16 f16;
typedef __attribute__((ext_vector_type(16))) _Float16 v16h;
typedef __attribute__((ext_vector_type(8)))  _Float16 v8h;
typedef __attribute__((ext_vector_type(8)))  float    v8f;

union AFrag {
    v16h v;
    v8h  h[2];
};

__global__ __launch_bounds__(256) void qtb_fused_kernel(
    const float* __restrict__ x,
    const float* __restrict__ attn_theta,
    const float* __restrict__ Wc,
    const float* __restrict__ bc,
    const float* __restrict__ g1,
    const float* __restrict__ beta1,
    const float* __restrict__ ffn_theta,
    const float* __restrict__ W1,
    const float* __restrict__ b1,
    const float* __restrict__ W2,
    const float* __restrict__ b2,
    const float* __restrict__ g2,
    const float* __restrict__ beta2,
    float* __restrict__ out)
{
    // 32KB + 16KB + 16KB = 64KB static LDS
    __shared__ __align__(16) f16   smH[64 * 256]; // q (first 8KB), later h [64][256]
    __shared__ __align__(16) f16   smW[64 * 128]; // WcT / W1t(padded) / W2t halves
    __shared__ __align__(16) float smY[64 * 64];  // x+attn -> y -> y+ffn

    const int tid  = threadIdx.x;
    const int lane = tid & 31;
    const int wv   = tid >> 5;          // wave id 0..7
    const int hf   = lane >> 4;         // fragment half (0/1)
    const int l16  = lane & 15;
    const int rowBase = blockIdx.x * 64;

    // ---------------- phase 0: q = qcircuit(x) -> smH[0:4096] f16; WcT -> smW
    {
        const float ct0 = __cosf(attn_theta[0]);
        const float t1  = attn_theta[1];
        const float ct3 = __cosf(attn_theta[3]);
        const float t4  = attn_theta[4];
#pragma unroll
        for (int i = 0; i < 2; ++i) {
            int id = tid + i * 256;           // 512 (row,group) tasks
            int r = id >> 3, g = id & 7;
            const float* xp = x + (size_t)(rowBase + r) * 64 + g * 8;
            float a0 = xp[0], a1 = xp[1], a2 = xp[2], a3 = xp[3];
            float a4 = xp[4], a5 = xp[5], a6 = xp[6], a7 = xp[7];
            float c0 = __cosf(a0), c1 = __cosf(a1);
            f16* qp = smH + r * 64 + g * 8;
            qp[0] = (f16)(ct0 * c0);
            qp[1] = (f16)c1;
            qp[2] = (f16)__cosf(a2 + t4);
            qp[3] = (f16)__cosf(a3 + t1);
            qp[4] = (f16)(c1 * __cosf(a4));
            qp[5] = (f16)__cosf(a5);
            qp[6] = (f16)(ct3 * __cosf(a6));
            qp[7] = (f16)(ct0 * c0 * __cosf(a7));
        }
#pragma unroll
        for (int i = 0; i < 16; ++i) {        // WcT: smW[n*64+k] = Wc[k][n]
            int id = tid + i * 256;           // 4096
            int n = id >> 6, k = id & 63;
            smW[n * 64 + k] = (f16)Wc[k * 64 + n];
        }
    }
    __syncthreads();

    // ---------------- phase 1: GEMM1  smY = x + (q @ Wc + bc)
    {
        const int m0 = (wv >> 1) * 16;        // tileM fixed per wave
        const int mrow = m0 + l16;
        AFrag a[2];
#pragma unroll
        for (int ks = 0; ks < 2; ++ks) {      // K = 0, 32
            const f16* p = smH + mrow * 64 + ks * 32 + hf * 8;
            a[ks].h[0] = *(const v8h*)p;
            a[ks].h[1] = *(const v8h*)(p + 16);
        }
#pragma unroll
        for (int i = 0; i < 2; ++i) {
            int tn = (wv & 1) * 2 + i;
            int ncol = tn * 16 + l16;
            v8f acc = {};
#pragma unroll
            for (int ks = 0; ks < 2; ++ks) {
                AFrag b;
                const f16* p = smW + ncol * 64 + ks * 32 + hf * 8;
                b.h[0] = *(const v8h*)p;
                b.h[1] = *(const v8h*)(p + 16);
                acc = __builtin_amdgcn_wmma_f32_16x16x32_f16(
                    false, a[ks].v, false, b.v, (short)0, acc, false, false);
            }
            float bias = bc[ncol];
#pragma unroll
            for (int v = 0; v < 8; ++v) {
                int m = m0 + v + hf * 8;
                smY[m * 64 + ncol] =
                    acc[v] + bias + x[(size_t)(rowBase + m) * 64 + ncol];
            }
        }
    }
    __syncthreads();

    // ---------------- phase 2: LN1 in place on smY
    {
        int r = tid >> 2, p = tid & 3;        // 4 lanes per row, in-wave
        float s = 0.f, ss = 0.f;
#pragma unroll
        for (int j = 0; j < 16; ++j) {
            float v = smY[r * 64 + p * 16 + j];
            s += v; ss += v * v;
        }
        s += __shfl_xor(s, 1); ss += __shfl_xor(ss, 1);
        s += __shfl_xor(s, 2); ss += __shfl_xor(ss, 2);
        float mean = s * (1.f / 64.f);
        float var  = ss * (1.f / 64.f) - mean * mean;
        float rstd = rsqrtf(var + 1e-5f);
#pragma unroll
        for (int j = 0; j < 16; ++j) {
            int n = p * 16 + j;
            float v = smY[r * 64 + n];
            smY[r * 64 + n] = (v - mean) * rstd * g1[n] + beta1[n];
        }
    }
    __syncthreads();

    // ---------------- phase 3: W1t (K padded 8->32) into smW [256][32]
#pragma unroll
    for (int i = 0; i < 32; ++i) {
        int id = tid + i * 256;               // 8192
        int n = id >> 5, k = id & 31;
        smW[n * 32 + k] = (k < 8) ? (f16)W1[k * 256 + n] : (f16)0.f;
    }
    __syncthreads();

    // ---------------- phase 4: GEMM2  h = relu(qf @ W1 + b1) -> smH [64][256]
    {
        const float ct0 = __cosf(ffn_theta[0]);
        const float t1  = ffn_theta[1];
        const float ct3 = __cosf(ffn_theta[3]);
        const float t4  = ffn_theta[4];
        const int m0 = (wv >> 1) * 16;
        const int mrow = m0 + l16;
        v16h av = {};                         // zero-padded A fragment
        if (hf == 0) {                        // only K=0..7 nonzero, lanes 0-15
            const float* yp = smY + mrow * 64;
            float y0 = yp[0], y1 = yp[1], y2 = yp[2], y3 = yp[3];
            float y4 = yp[4], y5 = yp[5], y6 = yp[6], y7 = yp[7];
            float c0 = __cosf(y0), c1 = __cosf(y1);
            av[0] = (f16)(ct0 * c0);
            av[1] = (f16)c1;
            av[2] = (f16)__cosf(y2 + t4);
            av[3] = (f16)__cosf(y3 + t1);
            av[4] = (f16)(c1 * __cosf(y4));
            av[5] = (f16)__cosf(y5);
            av[6] = (f16)(ct3 * __cosf(y6));
            av[7] = (f16)(ct0 * c0 * __cosf(y7));
        }
#pragma unroll
        for (int i = 0; i < 8; ++i) {
            int tn = (wv & 1) * 8 + i;        // 16 N tiles across 2 waves/tileM
            int ncol = tn * 16 + l16;
            AFrag b;
            const f16* p = smW + ncol * 32 + hf * 8;
            b.h[0] = *(const v8h*)p;
            b.h[1] = *(const v8h*)(p + 16);
            v8f acc = {};
            acc = __builtin_amdgcn_wmma_f32_16x16x32_f16(
                false, av, false, b.v, (short)0, acc, false, false);
            float bias = b1[ncol];
#pragma unroll
            for (int v = 0; v < 8; ++v) {
                int m = m0 + v + hf * 8;
                float hv = acc[v] + bias;
                smH[m * 256 + ncol] = (f16)fmaxf(hv, 0.f);
            }
        }
    }
    __syncthreads();

    // ---------------- phase 5: GEMM3  ffn = h @ W2 + b2 ; smY = y + ffn
    {
        const int m0 = (wv >> 1) * 16;
        const int mrow = m0 + l16;
        const int ncolA = ((wv & 1) * 2) * 16 + l16;
        const int ncolB = ncolA + 16;
        v8f accA = {}, accB = {};
#pragma unroll
        for (int kh = 0; kh < 2; ++kh) {      // W2 K-halves double-staged in smW
#pragma unroll
            for (int i = 0; i < 32; ++i) {    // load W2t half: smW[n*128+kk]
                int id = tid + i * 256;       // 8192
                int n = id >> 7, kk = id & 127;
                smW[n * 128 + kk] = (f16)W2[(size_t)(kh * 128 + kk) * 64 + n];
            }
            __syncthreads();
#pragma unroll
            for (int ks = 0; ks < 4; ++ks) {
                AFrag a;
                const f16* p = smH + mrow * 256 + kh * 128 + ks * 32 + hf * 8;
                a.h[0] = *(const v8h*)p;
                a.h[1] = *(const v8h*)(p + 16);
                AFrag bA, bB;
                const f16* qA = smW + ncolA * 128 + ks * 32 + hf * 8;
                bA.h[0] = *(const v8h*)qA;
                bA.h[1] = *(const v8h*)(qA + 16);
                const f16* qB = smW + ncolB * 128 + ks * 32 + hf * 8;
                bB.h[0] = *(const v8h*)qB;
                bB.h[1] = *(const v8h*)(qB + 16);
                accA = __builtin_amdgcn_wmma_f32_16x16x32_f16(
                    false, a.v, false, bA.v, (short)0, accA, false, false);
                accB = __builtin_amdgcn_wmma_f32_16x16x32_f16(
                    false, a.v, false, bB.v, (short)0, accB, false, false);
            }
            __syncthreads();                  // protect smW reload / smY rewrite
        }
        float valsA[8], valsB[8];
        float b2A = b2[ncolA], b2B = b2[ncolB];
#pragma unroll
        for (int v = 0; v < 8; ++v) {
            int m = m0 + v + hf * 8;
            valsA[v] = accA[v] + b2A + smY[m * 64 + ncolA];
            valsB[v] = accB[v] + b2B + smY[m * 64 + ncolB];
        }
        __syncthreads();                      // all y reads done before rewrite
#pragma unroll
        for (int v = 0; v < 8; ++v) {
            int m = m0 + v + hf * 8;
            smY[m * 64 + ncolA] = valsA[v];
            smY[m * 64 + ncolB] = valsB[v];
        }
    }
    __syncthreads();

    // ---------------- phase 6: LN2 and global store
    {
        int r = tid >> 2, p = tid & 3;
        float s = 0.f, ss = 0.f;
#pragma unroll
        for (int j = 0; j < 16; ++j) {
            float v = smY[r * 64 + p * 16 + j];
            s += v; ss += v * v;
        }
        s += __shfl_xor(s, 1); ss += __shfl_xor(ss, 1);
        s += __shfl_xor(s, 2); ss += __shfl_xor(ss, 2);
        float mean = s * (1.f / 64.f);
        float var  = ss * (1.f / 64.f) - mean * mean;
        float rstd = rsqrtf(var + 1e-5f);
        float* op = out + (size_t)(rowBase + r) * 64;
#pragma unroll
        for (int j = 0; j < 16; ++j) {
            int n = p * 16 + j;
            float v = smY[r * 64 + n];
            op[n] = (v - mean) * rstd * g2[n] + beta2[n];
        }
    }
}

extern "C" void kernel_launch(void* const* d_in, const int* in_sizes, int n_in,
                              void* d_out, int out_size, void* d_ws, size_t ws_size,
                              hipStream_t stream) {
    (void)n_in; (void)out_size; (void)d_ws; (void)ws_size;
    const float* x          = (const float*)d_in[0];
    const float* attn_theta = (const float*)d_in[1];
    const float* Wc         = (const float*)d_in[2];
    const float* bc         = (const float*)d_in[3];
    const float* g1         = (const float*)d_in[4];
    const float* beta1      = (const float*)d_in[5];
    const float* ffn_theta  = (const float*)d_in[6];
    const float* W1         = (const float*)d_in[7];
    const float* b1         = (const float*)d_in[8];
    const float* W2         = (const float*)d_in[9];
    const float* b2         = (const float*)d_in[10];
    const float* g2         = (const float*)d_in[11];
    const float* beta2      = (const float*)d_in[12];
    float* out = (float*)d_out;

    int rows = in_sizes[0] / 64;              // B*S = 4096
    int grid = rows / 64;                     // 64 rows per workgroup
    if (grid < 1) grid = 1;
    qtb_fused_kernel<<<grid, 256, 0, stream>>>(
        x, attn_theta, Wc, bc, g1, beta1, ffn_theta,
        W1, b1, W2, b2, g2, beta2, out);
}
